// HeteroGNN_53601191854187
// MI455X (gfx1250) — matrix operand order repaired and checked by reading, not compiled, not run on hardware
//
#include <hip/hip_runtime.h>
#include <hip/hip_bf16.h>

typedef __attribute__((ext_vector_type(16))) _Float16 v16h;
typedef __attribute__((ext_vector_type(8)))  _Float16 v8h;
typedef __attribute__((ext_vector_type(4)))  _Float16 v4h;
typedef __attribute__((ext_vector_type(8)))  float    v8f;

#define HDIM 256
#define NCELL 50000
#define NDRUG 10000

// ---------------- elementwise helpers ----------------

__global__ void k_f32_to_f16(const float* __restrict__ x, _Float16* __restrict__ y, int n) {
  for (int i = blockIdx.x * blockDim.x + threadIdx.x; i < n; i += gridDim.x * blockDim.x)
    y[i] = (_Float16)x[i];
}

// W: K x 256 row-major  ->  WT: 256 x K row-major (f16)
__global__ void k_transpose_f16(const float* __restrict__ W, _Float16* __restrict__ WT, int K) {
  int n = K * 256;
  for (int i = blockIdx.x * blockDim.x + threadIdx.x; i < n; i += gridDim.x * blockDim.x) {
    int k = i >> 8, c = i & 255;
    WT[(size_t)c * K + k] = (_Float16)W[i];
  }
}

__global__ void k_count(const int* __restrict__ dst, float* __restrict__ cnt, int E) {
  for (int i = blockIdx.x * blockDim.x + threadIdx.x; i < E; i += gridDim.x * blockDim.x)
    atomicAdd(&cnt[dst[i]], 1.0f);
}

// scatter-add: sums[dst[e]*F + f] += x[src[e]*F + f], 4 floats per thread
__global__ void k_aggregate_f32(const float* __restrict__ x, const int* __restrict__ src,
                                const int* __restrict__ dst, float* __restrict__ sums,
                                int E, int F) {
  int chunks = F >> 2;
  int total = E * chunks;
  for (int idx = blockIdx.x * blockDim.x + threadIdx.x; idx < total; idx += gridDim.x * blockDim.x) {
    int e = idx / chunks;
    int c = (idx - e * chunks) << 2;
    int s = src[e], d = dst[e];
    const float4 v = *(const float4*)(x + (size_t)s * F + c);
    float* o = sums + (size_t)d * F + c;
    atomicAdd(o + 0, v.x);
    atomicAdd(o + 1, v.y);
    atomicAdd(o + 2, v.z);
    atomicAdd(o + 3, v.w);
  }
}

__global__ void k_aggregate_f16(const _Float16* __restrict__ x, const int* __restrict__ src,
                                const int* __restrict__ dst, float* __restrict__ sums,
                                int E, int F) {
  int chunks = F >> 2;
  int total = E * chunks;
  for (int idx = blockIdx.x * blockDim.x + threadIdx.x; idx < total; idx += gridDim.x * blockDim.x) {
    int e = idx / chunks;
    int c = (idx - e * chunks) << 2;
    int s = src[e], d = dst[e];
    const v4h v = *(const v4h*)(x + (size_t)s * F + c);
    float* o = sums + (size_t)d * F + c;
    atomicAdd(o + 0, (float)v[0]);
    atomicAdd(o + 1, (float)v[1]);
    atomicAdd(o + 2, (float)v[2]);
    atomicAdd(o + 3, (float)v[3]);
  }
}

// mean = sums / max(cnt,1), converted to f16 (GEMM A operand)
__global__ void k_mean_f16(const float* __restrict__ sums, const float* __restrict__ cnt,
                           _Float16* __restrict__ out, int M, int F) {
  int n = M * F;
  for (int idx = blockIdx.x * blockDim.x + threadIdx.x; idx < n; idx += gridDim.x * blockDim.x) {
    int i = idx / F;
    float c = cnt[i];
    float d = c > 1.0f ? c : 1.0f;
    out[idx] = (_Float16)(sums[idx] / d);
  }
}

__global__ void k_relu_to_f16(const float* __restrict__ C, _Float16* __restrict__ out,
                              float scale, int n) {
  for (int i = blockIdx.x * blockDim.x + threadIdx.x; i < n; i += gridDim.x * blockDim.x) {
    float v = scale * C[i];
    out[i] = (_Float16)(v > 0.0f ? v : 0.0f);
  }
}

__global__ void k_relu_f32(const float* __restrict__ C, float* __restrict__ out,
                           float scale, int n) {
  for (int i = blockIdx.x * blockDim.x + threadIdx.x; i < n; i += gridDim.x * blockDim.x) {
    float v = scale * C[i];
    out[i] = v > 0.0f ? v : 0.0f;
  }
}

// ---------------- WMMA GEMM ----------------
// C[M x 256] (+)= A[M x K] @ W[K x 256]  (W given transposed: WT[256 x K], f16)
// One wave computes a 16x64 tile (four 16x16 WMMA accumulators).
// M % 16 == 0, K % 32 == 0 guaranteed by caller (NC=50000, ND=10000, K in {128,256}).
__global__ void k_gemm_wmma(const _Float16* __restrict__ A, const _Float16* __restrict__ WT,
                            const float* __restrict__ bias, float* __restrict__ C,
                            int M, int K, int accFlag) {
  int w    = (blockIdx.x * blockDim.x + threadIdx.x) >> 5;
  int lane = threadIdx.x & 31;
  int tiles = (M >> 4) * 4;            // (M/16) row tiles x (256/64) col groups
  if (w >= tiles) return;              // wave-uniform: EXEC stays all-1s
  int m0 = (w >> 2) << 4;
  int n0 = (w & 3) << 6;
  int lh = lane & 15;
  int lg = lane >> 4;

  union VA { v16h v; v8h h[2]; };

  v8f acc[4];
  for (int j = 0; j < 4; ++j) {
    int n = n0 + j * 16 + lh;
    float bv = bias ? bias[n] : 0.0f;
    for (int i = 0; i < 8; ++i) {
      int m = m0 + i + 8 * lg;
      float v = accFlag ? C[(size_t)m * HDIM + n] : 0.0f;
      acc[j][i] = v + bv;
    }
  }

  const _Float16* Arow = A + (size_t)(m0 + lh) * K;
  for (int k0 = 0; k0 < K; k0 += 32) {
    // A fragment (16x32 f16): lane holds row m0+lh;
    // halves 0..7 = K k0+16*lg .. +7 ; halves 8..15 = K k0+16*lg+8 .. +7
    VA a;
    a.h[0] = *(const v8h*)(Arow + k0 + 16 * lg);
    a.h[1] = *(const v8h*)(Arow + k0 + 16 * lg + 8);
    for (int j = 0; j < 4; ++j) {
      // B fragment (32x16 f16): lane holds column n; 16 consecutive K starting k0+16*lg
      const _Float16* Bp = WT + (size_t)(n0 + j * 16 + lh) * K + k0 + 16 * lg;
      VA b;
      b.h[0] = *(const v8h*)(Bp);
      b.h[1] = *(const v8h*)(Bp + 8);
      acc[j] = __builtin_amdgcn_wmma_f32_16x16x32_f16(
          false, a.v, false, b.v, (short)0, acc[j], false, false);
    }
  }

  for (int j = 0; j < 4; ++j) {
    int n = n0 + j * 16 + lh;
    for (int i = 0; i < 8; ++i) {
      int m = m0 + i + 8 * lg;
      C[(size_t)m * HDIM + n] = acc[j][i];
    }
  }
}

// ---------------- host orchestration ----------------

static inline int cdiv(int a, int b) { return (a + b - 1) / b; }

extern "C" void kernel_launch(void* const* d_in, const int* in_sizes, int n_in,
                              void* d_out, int out_size, void* d_ws, size_t ws_size,
                              hipStream_t stream) {
  const float* x_cell = (const float*)d_in[0];
  const float* x_drug = (const float*)d_in[1];
  const int* resp_src = (const int*)d_in[2];
  const int* resp_dst = (const int*)d_in[3];
  const int* cls_src  = (const int*)d_in[4];
  const int* cls_dst  = (const int*)d_in[5];
  const int* drs_src  = (const int*)d_in[6];
  const int* drs_dst  = (const int*)d_in[7];
  const int E_resp = in_sizes[2], E_cls = in_sizes[4], E_drs = in_sizes[6];

  // weights (f32), order per setup_inputs
  const float* W[12];   // W1l W1r W2l W2r W3l W3r V1l V1r V2l V2r V3l V3r
  const float* bsrc[6]; // b1l b2l b3l c1l c2l c3l
  W[0] = (const float*)d_in[8];   bsrc[0] = (const float*)d_in[9];   W[1] = (const float*)d_in[10];
  W[2] = (const float*)d_in[11];  bsrc[1] = (const float*)d_in[12];  W[3] = (const float*)d_in[13];
  W[4] = (const float*)d_in[14];  bsrc[2] = (const float*)d_in[15];  W[5] = (const float*)d_in[16];
  W[6] = (const float*)d_in[17];  bsrc[3] = (const float*)d_in[18];  W[7] = (const float*)d_in[19];
  W[8] = (const float*)d_in[20];  bsrc[4] = (const float*)d_in[21];  W[9] = (const float*)d_in[22];
  W[10] = (const float*)d_in[23]; bsrc[5] = (const float*)d_in[24];  W[11] = (const float*)d_in[25];
  const int WK[12] = {256, 128, 256, 256, 128, 128, 256, 256, 256, 256, 256, 256};

  // ---- workspace layout (256B aligned slabs) ----
  char* ws = (char*)d_ws;
  size_t off = 0;
  auto alloc = [&](size_t bytes) -> char* {
    char* p = ws + off;
    off = (off + bytes + 255) & ~(size_t)255;
    return p;
  };
  float*    cntResp = (float*)alloc((size_t)NDRUG * 4);
  float*    cntCls  = (float*)alloc((size_t)NCELL * 4);
  float*    cntDrs  = (float*)alloc((size_t)NDRUG * 4);
  float*    sumResp = (float*)alloc((size_t)NDRUG * 256 * 4);  // aliased as Cdrug
  float*    sumCls  = (float*)alloc((size_t)NCELL * 256 * 4);  // aliased as Ccell
  float*    sumDrs  = (float*)alloc((size_t)NDRUG * 256 * 4);
  _Float16* mResp   = (_Float16*)alloc((size_t)NDRUG * 256 * 2);
  _Float16* mCls    = (_Float16*)alloc((size_t)NCELL * 256 * 2);
  _Float16* mDrs    = (_Float16*)alloc((size_t)NDRUG * 256 * 2);
  _Float16* xcH     = (_Float16*)alloc((size_t)NCELL * 256 * 2); // becomes hcH after layer 1
  _Float16* xdH     = (_Float16*)alloc((size_t)NDRUG * 256 * 2); // becomes hdH after layer 1
  _Float16* WT[12];
  for (int i = 0; i < 12; ++i) WT[i] = (_Float16*)alloc((size_t)256 * 256 * 2);
  float* Ccell = sumCls;   // sums consumed into mCls before GEMMs write here
  float* Cdrug = sumResp;  // sums consumed into mResp before GEMMs write here
  (void)ws_size;

  const int TB = 256;
  const int GRID_BIG = 4096;

  // ---- weight transpose-convert + input f16 copies ----
  for (int i = 0; i < 12; ++i)
    k_transpose_f16<<<cdiv(WK[i] * 256, TB), TB, 0, stream>>>(W[i], WT[i], WK[i]);
  k_f32_to_f16<<<GRID_BIG, TB, 0, stream>>>(x_cell, xcH, NCELL * 256);
  k_f32_to_f16<<<cdiv(NDRUG * 128, TB), TB, 0, stream>>>(x_drug, xdH, NDRUG * 128);

  // ---- degree counts (same for both layers) ----
  hipMemsetAsync(cntResp, 0, (size_t)NDRUG * 4, stream);
  hipMemsetAsync(cntCls, 0, (size_t)NCELL * 4, stream);
  hipMemsetAsync(cntDrs, 0, (size_t)NDRUG * 4, stream);
  k_count<<<cdiv(E_resp, TB), TB, 0, stream>>>(resp_dst, cntResp, E_resp);
  k_count<<<cdiv(E_cls, TB), TB, 0, stream>>>(cls_dst, cntCls, E_cls);
  k_count<<<cdiv(E_drs, TB), TB, 0, stream>>>(drs_dst, cntDrs, E_drs);

  // ================= layer 1 =================
  hipMemsetAsync(sumResp, 0, (size_t)NDRUG * 256 * 4, stream);
  hipMemsetAsync(sumCls, 0, (size_t)NCELL * 256 * 4, stream);
  hipMemsetAsync(sumDrs, 0, (size_t)NDRUG * 256 * 4, stream);
  k_aggregate_f32<<<GRID_BIG, TB, 0, stream>>>(x_cell, resp_src, resp_dst, sumResp, E_resp, 256);
  k_aggregate_f32<<<GRID_BIG, TB, 0, stream>>>(x_cell, cls_src, cls_dst, sumCls, E_cls, 256);
  k_aggregate_f32<<<GRID_BIG, TB, 0, stream>>>(x_drug, drs_src, drs_dst, sumDrs, E_drs, 128);
  k_mean_f16<<<cdiv(NDRUG * 256, TB), TB, 0, stream>>>(sumResp, cntResp, mResp, NDRUG, 256);
  k_mean_f16<<<GRID_BIG, TB, 0, stream>>>(sumCls, cntCls, mCls, NCELL, 256);
  k_mean_f16<<<cdiv(NDRUG * 128, TB), TB, 0, stream>>>(sumDrs, cntDrs, mDrs, NDRUG, 128);

  const int gCell = cdiv((NCELL / 16) * 4, 8);  // 8 waves/block
  const int gDrug = cdiv((NDRUG / 16) * 4, 8);
  // cell: relu(mean_cls@W2l + b2l + x_cell@W2r)
  k_gemm_wmma<<<gCell, TB, 0, stream>>>(mCls, WT[2], bsrc[1], Ccell, NCELL, 256, 0);
  k_gemm_wmma<<<gCell, TB, 0, stream>>>(xcH, WT[3], nullptr, Ccell, NCELL, 256, 1);
  // drug: relu(0.5*(mean_resp@W1l + b1l + x_drug@W1r + mean_drs@W3l + b3l + x_drug@W3r))
  k_gemm_wmma<<<gDrug, TB, 0, stream>>>(mResp, WT[0], bsrc[0], Cdrug, NDRUG, 256, 0);
  k_gemm_wmma<<<gDrug, TB, 0, stream>>>(xdH, WT[1], nullptr, Cdrug, NDRUG, 128, 1);
  k_gemm_wmma<<<gDrug, TB, 0, stream>>>(mDrs, WT[4], bsrc[2], Cdrug, NDRUG, 128, 1);
  k_gemm_wmma<<<gDrug, TB, 0, stream>>>(xdH, WT[5], nullptr, Cdrug, NDRUG, 128, 1);

  _Float16* hcH = xcH;
  _Float16* hdH = xdH;
  k_relu_to_f16<<<GRID_BIG, TB, 0, stream>>>(Ccell, hcH, 1.0f, NCELL * 256);
  k_relu_to_f16<<<cdiv(NDRUG * 256, TB), TB, 0, stream>>>(Cdrug, hdH, 0.5f, NDRUG * 256);

  // ================= layer 2 =================
  hipMemsetAsync(sumResp, 0, (size_t)NDRUG * 256 * 4, stream);
  hipMemsetAsync(sumCls, 0, (size_t)NCELL * 256 * 4, stream);
  hipMemsetAsync(sumDrs, 0, (size_t)NDRUG * 256 * 4, stream);
  k_aggregate_f16<<<GRID_BIG, TB, 0, stream>>>(hcH, resp_src, resp_dst, sumResp, E_resp, 256);
  k_aggregate_f16<<<GRID_BIG, TB, 0, stream>>>(hcH, cls_src, cls_dst, sumCls, E_cls, 256);
  k_aggregate_f16<<<GRID_BIG, TB, 0, stream>>>(hdH, drs_src, drs_dst, sumDrs, E_drs, 256);
  k_mean_f16<<<cdiv(NDRUG * 256, TB), TB, 0, stream>>>(sumResp, cntResp, mResp, NDRUG, 256);
  k_mean_f16<<<GRID_BIG, TB, 0, stream>>>(sumCls, cntCls, mCls, NCELL, 256);
  k_mean_f16<<<cdiv(NDRUG * 256, TB), TB, 0, stream>>>(sumDrs, cntDrs, mDrs, NDRUG, 256);

  // cell: relu(mean_cls@V2l + c2l + hc@V2r)
  k_gemm_wmma<<<gCell, TB, 0, stream>>>(mCls, WT[8], bsrc[4], Ccell, NCELL, 256, 0);
  k_gemm_wmma<<<gCell, TB, 0, stream>>>(hcH, WT[9], nullptr, Ccell, NCELL, 256, 1);
  // drug: relu(0.5*(mean_resp@V1l + c1l + hd@V1r + mean_drs@V3l + c3l + hd@V3r))
  k_gemm_wmma<<<gDrug, TB, 0, stream>>>(mResp, WT[6], bsrc[3], Cdrug, NDRUG, 256, 0);
  k_gemm_wmma<<<gDrug, TB, 0, stream>>>(hdH, WT[7], nullptr, Cdrug, NDRUG, 256, 1);
  k_gemm_wmma<<<gDrug, TB, 0, stream>>>(mDrs, WT[10], bsrc[5], Cdrug, NDRUG, 256, 1);
  k_gemm_wmma<<<gDrug, TB, 0, stream>>>(hdH, WT[11], nullptr, Cdrug, NDRUG, 256, 1);

  // final relu -> d_out (cell block then drug block, f32)
  float* outCell = (float*)d_out;
  float* outDrug = outCell + (size_t)NCELL * 256;
  k_relu_f32<<<GRID_BIG, TB, 0, stream>>>(Ccell, outCell, 1.0f, NCELL * 256);
  k_relu_f32<<<cdiv(NDRUG * 256, TB), TB, 0, stream>>>(Cdrug, outDrug, 0.5f, NDRUG * 256);
}
